// SequenceAttention_59459527246270
// MI455X (gfx1250) — matrix operand order repaired and checked by reading, not compiled, not run on hardware
//
#include <hip/hip_runtime.h>

// ---------------------------------------------------------------------------
// SequenceAttention for MI455X (gfx1250, wave32, WMMA bf16)
//   q = relu(input @ W^T + b);  k = relu(context @ W^T + b)
//   out = softmax(mask(q @ k^T)) @ k
// Matmuls on v_wmma_f32_16x16x32_bf16 (f32 accumulate).
// Projection GEMM stages W tiles in LDS with global_load_async_to_lds_b128
// (ASYNCcnt) double buffering; attention loops are software-pipelined.
// ---------------------------------------------------------------------------

typedef __bf16 bf16_t;
typedef __attribute__((ext_vector_type(16))) __bf16 v16bf;
typedef __attribute__((ext_vector_type(8)))  __bf16 v8bf;
typedef __attribute__((ext_vector_type(8)))  float  v8f;

static constexpr int B_  = 16;
static constexpr int LI_ = 2048;
static constexpr int LC_ = 2048;
static constexpr int H_  = 1024;
static constexpr float NEG_BIG = -3.0e38f;

__device__ __forceinline__ v16bf cat8(v8bf lo, v8bf hi) {
    return __builtin_shufflevector(lo, hi, 0,1,2,3,4,5,6,7,8,9,10,11,12,13,14,15);
}

__device__ __forceinline__ v8f wmma_bf16(v16bf a, v16bf b, v8f c) {
    // (neg_a, A, neg_b, B, c_mod, C, reuse_a, reuse_b)
    return __builtin_amdgcn_wmma_f32_16x16x32_bf16(false, a, false, b, (short)0, c,
                                                   false, false);
}

// ---------------------------------------------------------------------------
// Kernel 1: W fp32 -> bf16 (vectorized x8)
// ---------------------------------------------------------------------------
__global__ __launch_bounds__(256)
void convert_w_kernel(const float* __restrict__ W, bf16_t* __restrict__ Wb) {
    int idx = (blockIdx.x * 256 + threadIdx.x) * 8;
    v8f f = *(const v8f*)(W + idx);
    v8bf h;
    #pragma unroll
    for (int e = 0; e < 8; ++e) h[e] = (bf16_t)f[e];
    *(v8bf*)(Wb + idx) = h;
}

// ---------------------------------------------------------------------------
// Kernel 2: projection GEMM + bias + relu.
//   out[row][n] = relu(sum_h src[row][h] * W[n][h] + b[n])
// Block = 8 waves => 64(M) x 128(N); per wave 16x64.
// W tile (128 rows x 32 k, bf16) staged in LDS, double-buffered through
// global_load_async_to_lds_b128 / s_wait_asynccnt. Row stride padded to 40
// bf16 (80 B) so the 16-lane ds_load_b128 pattern is bank-conflict-free.
// ---------------------------------------------------------------------------
__global__ __launch_bounds__(256)
void proj_relu_kernel(const float* __restrict__ src, const float* __restrict__ bias,
                      const bf16_t* __restrict__ Wb,
                      bf16_t* __restrict__ out_rm, bf16_t* __restrict__ out_tr) {
    constexpr int WROW = 40;                 // padded LDS row stride (bf16 elems)

    const int lane = threadIdx.x & 31;
    const int wave = threadIdx.x >> 5;
    const int hi   = lane >> 4;              // half-wave: 0 or 1
    const int r    = lane & 15;
    const int wm   = wave >> 1, wn = wave & 1;
    const int rowbase = blockIdx.x * 64 + wm * 16;
    const int nblk    = blockIdx.y * 128;
    const int n0      = nblk + wn * 64;

    __shared__ bf16_t wtile[2][128 * WROW];  // 2 x 10 KB

    // Cooperative async stage: 128 rows x 32 bf16 = 8 KB = 512 x 16 B chunks;
    // 256 threads x 2 chunks, each lane one async b128 into LDS.
    auto issue_stage = [&](int buf, int kk) {
        #pragma unroll
        for (int s = 0; s < 2; ++s) {
            const int c  = threadIdx.x + 256 * s;
            const int ln = c >> 2;              // local W row 0..127
            const int co = (c & 3) * 8;         // bf16 offset within row
            const bf16_t* gsrc = Wb + (size_t)(nblk + ln) * H_ + kk + co;
            unsigned int ldst =
                (unsigned int)(uintptr_t)&wtile[buf][ln * WROW + co];
            asm volatile("global_load_async_to_lds_b128 %0, %1, off"
                         :: "v"(ldst), "v"(gsrc) : "memory");
        }
    };

    v8f acc[4] = {};

    // A operand (16x32 bf16): lane(lo) row r holds K={kk+0..7, kk+16..23};
    // lane(hi) holds K={kk+8..15, kk+24..31}.
    const float* arow = src + (size_t)(rowbase + r) * H_ + 8 * hi;

    issue_stage(0, 0);
    v8f f0 = *(const v8f*)(arow);
    v8f f1 = *(const v8f*)(arow + 16);

    int p = 0;
    for (int kk = 0; kk < H_; kk += 32) {
        const bool more = (kk + 32) < H_;
        if (more) {
            issue_stage(p ^ 1, kk + 32);
            asm volatile("s_wait_asynccnt 2" ::: "memory");  // stage p landed
        } else {
            asm volatile("s_wait_asynccnt 0" ::: "memory");
        }

        // convert current A fragment while waiting
        v16bf A;
        #pragma unroll
        for (int e = 0; e < 8; ++e) { A[e] = (bf16_t)f0[e]; A[8 + e] = (bf16_t)f1[e]; }

        __syncthreads();                       // wtile[p] visible to all waves

        if (more) {                            // prefetch next A (fp32)
            f0 = *(const v8f*)(arow + kk + 32);
            f1 = *(const v8f*)(arow + kk + 48);
        }

        #pragma unroll
        for (int nt = 0; nt < 4; ++nt) {
            // B operand: lane holds column n = wn*64+nt*16+r, K={8hi..8hi+7, 16+8hi..16+8hi+7}
            const bf16_t* wp = &wtile[p][(wn * 64 + nt * 16 + r) * WROW + 8 * hi];
            acc[nt] = wmma_bf16(A, cat8(*(const v8bf*)wp, *(const v8bf*)(wp + 16)),
                                acc[nt]);
        }
        __syncthreads();                       // wtile[p] free for next issue
        p ^= 1;
    }

    const int bb = rowbase >> 11;              // batch (rows per batch = 2048)
    const int jb = rowbase & (LC_ - 1);        // row within batch

    #pragma unroll
    for (int nt = 0; nt < 4; ++nt) {
        const int n = n0 + nt * 16 + r;
        const float bv = bias[n];
        bf16_t h[8];
        #pragma unroll
        for (int v = 0; v < 8; ++v) h[v] = (bf16_t)fmaxf(acc[nt][v] + bv, 0.0f);

        // C layout: lane holds column n; VGPR v is row m = v + 8*hi
        #pragma unroll
        for (int v = 0; v < 8; ++v) {
            const int row = rowbase + v + 8 * hi;
            out_rm[(size_t)row * H_ + n] = h[v];
        }
        if (out_tr) {
            // transposed element (n, j=jb+m): consecutive m -> contiguous b128
            v8bf pack;
            #pragma unroll
            for (int v = 0; v < 8; ++v) pack[v] = h[v];
            *(v8bf*)(out_tr + (size_t)bb * H_ * LC_ + (size_t)n * LC_ + jb + 8 * hi) = pack;
        }
    }
}

// ---------------------------------------------------------------------------
// Kernel 3: masked flash attention with V = k.
// One workgroup (16 waves, 512 thr) per (batch, 16-row q tile).
// Wave w owns output dims [64w, 64w+64). Both WMMA loops software-pipelined.
// ---------------------------------------------------------------------------
__global__ __launch_bounds__(512)
void attn_kernel(const bf16_t* __restrict__ q, const bf16_t* __restrict__ k,
                 const bf16_t* __restrict__ kT, const unsigned char* __restrict__ mask,
                 float* __restrict__ out) {
    constexpr int TJ = 256;
    constexpr int NW = 16;

    const int lane = threadIdx.x & 31;
    const int wave = threadIdx.x >> 5;
    const int hi   = lane >> 4;
    const int r    = lane & 15;
    const int b    = blockIdx.y;
    const int i0   = blockIdx.x * 16;
    const int n0   = wave * 64;

    __shared__ bf16_t p_buf[16][TJ];           // P tile, row-major, 8 KB
    __shared__ float  wmax[NW][16];
    __shared__ float  wsum[NW][16];
    __shared__ float  Mrow[16], Lrow[16], Srow[16];

    if (wave == 0 && lane < 16) { Mrow[lane] = NEG_BIG; Lrow[lane] = 0.0f; Srow[lane] = 0.0f; }

    v8f O[4] = {};

    const bf16_t* qrow = q + (size_t)(b * LI_ + i0 + r) * H_ + 8 * hi;  // A base
    const bf16_t* kTb  = kT + (size_t)b * H_ * LC_;

    __syncthreads();

    for (int it = 0; it < LC_ / TJ; ++it) {
        const int j0 = it * TJ;
        const int jw = j0 + wave * 16;

        // ---- scores: S(16x16) = q(16x1024) . k[jw..jw+16)^T, pipelined ----
        v8f S = {};
        const bf16_t* krow = k + (size_t)(b * LC_ + jw + r) * H_ + 8 * hi;
        v8bf qa0 = *(const v8bf*)(qrow);
        v8bf qa1 = *(const v8bf*)(qrow + 16);
        v8bf kb0 = *(const v8bf*)(krow);
        v8bf kb1 = *(const v8bf*)(krow + 16);
        #pragma unroll 4
        for (int kk = 0; kk < H_ - 32; kk += 32) {
            v8bf nq0 = *(const v8bf*)(qrow + kk + 32);
            v8bf nq1 = *(const v8bf*)(qrow + kk + 48);
            v8bf nk0 = *(const v8bf*)(krow + kk + 32);
            v8bf nk1 = *(const v8bf*)(krow + kk + 48);
            S = wmma_bf16(cat8(qa0, qa1), cat8(kb0, kb1), S);
            qa0 = nq0; qa1 = nq1; kb0 = nk0; kb1 = nk1;
        }
        S = wmma_bf16(cat8(qa0, qa1), cat8(kb0, kb1), S);

        // ---- mask (lane's column is j = jw + r for all 8 rows it holds) ----
        if (mask[b * LC_ + jw + r]) {
            #pragma unroll
            for (int v = 0; v < 8; ++v) S[v] = NEG_BIG;
        }

        // ---- local row max over 16 columns (within half-wave) ----
        #pragma unroll
        for (int v = 0; v < 8; ++v) {
            float x = S[v];
            x = fmaxf(x, __shfl_xor(x, 1));
            x = fmaxf(x, __shfl_xor(x, 2));
            x = fmaxf(x, __shfl_xor(x, 4));
            x = fmaxf(x, __shfl_xor(x, 8));
            if (r == 0) wmax[wave][v + 8 * hi] = x;
        }
        __syncthreads();

        // ---- wave0 combines: new running max + rescale factor ----
        if (wave == 0 && lane < 16) {
            float tm = wmax[0][lane];
            #pragma unroll
            for (int w = 1; w < NW; ++w) tm = fmaxf(tm, wmax[w][lane]);
            float Mo = Mrow[lane];
            float Mn = fmaxf(Mo, tm);
            Srow[lane] = __expf(Mo - Mn);
            Mrow[lane] = Mn;
        }
        __syncthreads();

        // ---- P = exp(S - Mn); partial sums; stage P; rescale O ----
        float Mn8[8], sc8[8];
        #pragma unroll
        for (int v = 0; v < 8; ++v) { Mn8[v] = Mrow[v + 8 * hi]; sc8[v] = Srow[v + 8 * hi]; }

        #pragma unroll
        for (int v = 0; v < 8; ++v) {
            float p = __expf(S[v] - Mn8[v]);
            p_buf[v + 8 * hi][wave * 16 + r] = (bf16_t)p;
            float s = p;
            s += __shfl_xor(s, 1);
            s += __shfl_xor(s, 2);
            s += __shfl_xor(s, 4);
            s += __shfl_xor(s, 8);
            if (r == 0) wsum[wave][v + 8 * hi] = s;
        }
        #pragma unroll
        for (int nt = 0; nt < 4; ++nt)
            #pragma unroll
            for (int v = 0; v < 8; ++v) O[nt][v] *= sc8[v];
        __syncthreads();

        // ---- wave0 updates running denominator ----
        if (wave == 0 && lane < 16) {
            float s = 0.0f;
            #pragma unroll
            for (int w = 0; w < NW; ++w) s += wsum[w][lane];
            Lrow[lane] = Lrow[lane] * Srow[lane] + s;
        }

        // ---- O += P(16x256) @ kT-slice(256x64), pipelined ----
        v8bf pa0 = *(const v8bf*)&p_buf[r][16 * hi];
        v8bf pa1 = *(const v8bf*)&p_buf[r][16 * hi + 8];
        v8bf vb0[4], vb1[4];
        #pragma unroll
        for (int nt = 0; nt < 4; ++nt) {
            const bf16_t* kp = kTb + (size_t)(n0 + nt * 16 + r) * LC_ + j0 + 16 * hi;
            vb0[nt] = *(const v8bf*)kp;
            vb1[nt] = *(const v8bf*)(kp + 8);
        }
        #pragma unroll
        for (int kk = 0; kk < TJ - 32; kk += 32) {
            v8bf na0 = *(const v8bf*)&p_buf[r][kk + 32 + 16 * hi];
            v8bf na1 = *(const v8bf*)&p_buf[r][kk + 32 + 16 * hi + 8];
            v8bf nb0[4], nb1[4];
            #pragma unroll
            for (int nt = 0; nt < 4; ++nt) {
                const bf16_t* kp = kTb + (size_t)(n0 + nt * 16 + r) * LC_
                                       + j0 + kk + 32 + 16 * hi;
                nb0[nt] = *(const v8bf*)kp;
                nb1[nt] = *(const v8bf*)(kp + 8);
            }
            v16bf A = cat8(pa0, pa1);
            #pragma unroll
            for (int nt = 0; nt < 4; ++nt)
                O[nt] = wmma_bf16(A, cat8(vb0[nt], vb1[nt]), O[nt]);
            pa0 = na0; pa1 = na1;
            #pragma unroll
            for (int nt = 0; nt < 4; ++nt) { vb0[nt] = nb0[nt]; vb1[nt] = nb1[nt]; }
        }
        {
            v16bf A = cat8(pa0, pa1);
            #pragma unroll
            for (int nt = 0; nt < 4; ++nt)
                O[nt] = wmma_bf16(A, cat8(vb0[nt], vb1[nt]), O[nt]);
        }
        __syncthreads();  // protect p_buf/wmax/Srow for next iteration
    }

    // ---- normalize & store fp32 output ----
    float l8[8];
    #pragma unroll
    for (int v = 0; v < 8; ++v) l8[v] = Lrow[v + 8 * hi];
    #pragma unroll
    for (int nt = 0; nt < 4; ++nt) {
        const int n = n0 + nt * 16 + r;
        #pragma unroll
        for (int v = 0; v < 8; ++v) {
            const int row = i0 + v + 8 * hi;
            out[(size_t)(b * LI_ + row) * H_ + n] = O[nt][v] / l8[v];
        }
    }
}

// ---------------------------------------------------------------------------
// Launch
// ---------------------------------------------------------------------------
extern "C" void kernel_launch(void* const* d_in, const int* in_sizes, int n_in,
                              void* d_out, int out_size, void* d_ws, size_t ws_size,
                              hipStream_t stream) {
    const float*         inp   = (const float*)d_in[0];
    const float*         ctx   = (const float*)d_in[1];
    const unsigned char* cmask = (const unsigned char*)d_in[2];  // jax bool = 1 byte
    const float*         W     = (const float*)d_in[3];
    const float*         bias  = (const float*)d_in[4];
    float*               out   = (float*)d_out;

    char* ws = (char*)d_ws;
    bf16_t* Wb  = (bf16_t*)(ws);                               //   2 MB
    bf16_t* qb  = (bf16_t*)(ws + (size_t)2   * 1024 * 1024);   //  64 MB
    bf16_t* kb  = (bf16_t*)(ws + (size_t)66  * 1024 * 1024);   //  64 MB
    bf16_t* kTb = (bf16_t*)(ws + (size_t)130 * 1024 * 1024);   //  64 MB

    convert_w_kernel<<<dim3((H_ * H_) / (256 * 8)), dim3(256), 0, stream>>>(W, Wb);

    proj_relu_kernel<<<dim3((B_ * LI_) / 64, H_ / 128), dim3(256), 0, stream>>>(
        inp, bias, Wb, qb, (bf16_t*)nullptr);
    proj_relu_kernel<<<dim3((B_ * LC_) / 64, H_ / 128), dim3(256), 0, stream>>>(
        ctx, bias, Wb, kb, kTb);

    attn_kernel<<<dim3(LI_ / 16, B_), dim3(512), 0, stream>>>(qb, kb, kTb, cmask, out);
}
